// ImprovedConvLayer_27633819583186
// MI455X (gfx1250) — matrix operand order repaired and checked by reading, not compiled, not run on hardware
//
#include <hip/hip_runtime.h>
#include <hip/hip_bf16.h>
#include <math.h>

#define NATOM 8000
#define MNBR 12
#define AF 64
#define KTOT 192            // 2A+F
#define COUT 512            // G(128) | E(128) | Pj(128) | Pl(128)
#define NROWS (NATOM * MNBR) // 96000
#define BN_EPS 1e-5f

typedef __bf16 bf16_t;
typedef __bf16 v8bf  __attribute__((ext_vector_type(8)));
typedef __bf16 v16bf __attribute__((ext_vector_type(16)));
typedef float  v8f   __attribute__((ext_vector_type(8)));

__device__ __forceinline__ float sigmoid_f(float x) { return 1.0f / (1.0f + __expf(-x)); }
__device__ __forceinline__ float softplus_f(float x) { return x > 20.0f ? x : log1pf(__expf(x)); }

// ---------------------------------------------------------------------------
// Build combined bf16 weight matrix Wcat (192 x 512) and bias (512):
//   cols [0,128)   = W_full            cols [128,256) = W_edge
//   cols [256,384) = [0; Wj; Wij]      cols [384,512) = [0; Wl; Wil]
// ---------------------------------------------------------------------------
__global__ __launch_bounds__(256) void k_prep_weights(
    const float* __restrict__ Wf, const float* __restrict__ We,
    const float* __restrict__ W3, const float* __restrict__ bfull,
    const float* __restrict__ bedge, bf16_t* __restrict__ Wcat,
    float* __restrict__ biascat)
{
    int i = blockIdx.x * 256 + threadIdx.x;
    if (i < KTOT * COUT) {
        int k = i / COUT, c = i - k * COUT;
        float w;
        if (c < 128)      w = Wf[k * 128 + c];
        else if (c < 256) w = We[k * 128 + (c - 128)];
        else if (c < 384) {
            int cc = c - 256;   // Pj: rows 64..127 -> Wj (W3 row k), rows 128..191 -> Wij (W3 row k+64)
            w = (k < 64) ? 0.0f : (k < 128 ? W3[k * 128 + cc] : W3[(k + 64) * 128 + cc]);
        } else {
            int cc = c - 384;   // Pl: rows 64..127 -> Wl (W3 row k+64), rows 128..191 -> Wil (W3 row k+128)
            w = (k < 64) ? 0.0f : (k < 128 ? W3[(k + 64) * 128 + cc] : W3[(k + 128) * 128 + cc]);
        }
        Wcat[i] = (bf16_t)w;
    }
    if (i < COUT)
        biascat[i] = (i < 128) ? bfull[i] : (i < 256 ? bedge[i - 128] : 0.0f);
}

// ---------------------------------------------------------------------------
// Pi[n][c] = atom_in[n] . Wa[:,c] + b_3body[c]   (8000 x 128, tiny GEMM)
// ---------------------------------------------------------------------------
__global__ __launch_bounds__(128) void k_pi(
    const float* __restrict__ atom_in, const float* __restrict__ W3,
    const float* __restrict__ b3, float* __restrict__ Pi)
{
    __shared__ float xs[64];
    int n = blockIdx.x, c = threadIdx.x;
    if (c < 64) xs[c] = atom_in[n * 64 + c];
    __syncthreads();
    float acc = b3[c];
#pragma unroll 8
    for (int k = 0; k < 64; k++) acc += xs[k] * W3[k * 128 + c];
    Pi[n * 128 + c] = acc;
}

// ---------------------------------------------------------------------------
// Fused GEMM: act(96000 x 512) = total(96000 x 192) @ Wcat(192 x 512) + bias
// total rows built on the fly ( [x_i | gather(x) | e_ij] -> bf16 in LDS ).
// 8 waves/block, wave = one 16x16 f32 tile, v_wmma_f32_16x16x32_bf16, 6 K-steps.
// ---------------------------------------------------------------------------
__global__ __launch_bounds__(256) void k_gemm(
    const float* __restrict__ atom_in, const float* __restrict__ nbr,
    const int* __restrict__ idx, const bf16_t* __restrict__ Wcat,
    const float* __restrict__ biascat, float* __restrict__ act)
{
    __shared__ bf16_t ldsA[64 * KTOT];   // 24.0 KB
    __shared__ bf16_t ldsB[32 * KTOT];   // 12.0 KB
    const int tid = threadIdx.x;
    const int rowBase = blockIdx.x * 64;
    const int colBase = blockIdx.y * 32;

    // stage A tile (build "total" rows, f32 -> bf16)
    for (int i = tid; i < 64 * KTOT; i += 256) {
        int r = i / KTOT, k = i - r * KTOT;
        int row = rowBase + r;
        int n = row / MNBR;
        float v;
        if (k < 64)       v = atom_in[n * 64 + k];
        else if (k < 128) v = atom_in[idx[row] * 64 + (k - 64)];
        else              v = nbr[row * 64 + (k - 128)];
        ldsA[i] = (bf16_t)v;
    }
    // stage B tile transposed: ldsB[col][k]
    for (int i = tid; i < 32 * KTOT; i += 256) {
        int cl = i / KTOT, k = i - cl * KTOT;
        ldsB[i] = Wcat[k * COUT + colBase + cl];
    }
    __syncthreads();

    const int lane = tid & 31;
    const int wave = tid >> 5;
    const int wrow = (wave & 3) * 16;   // 4 row sub-tiles
    const int wcol = (wave >> 2) * 16;  // 2 col sub-tiles
    const int lm = lane & 15;
    const int lh = lane >> 4;

    v8f acc = {};
#pragma unroll
    for (int ks = 0; ks < 6; ks++) {
        const int kb = ks * 32;
        // A fragment: lane row m=lm; 16 contiguous K per lane (lanes>=16 -> K+16)
        const v16bf a = *(const v16bf*)&ldsA[(wrow + lm) * KTOT + kb + lh * 16];
        // B fragment: lane col n=lm; 16 contiguous K per lane (lanes>=16 -> K+16)
        const v16bf b = *(const v16bf*)&ldsB[(wcol + lm) * KTOT + kb + lh * 16];
        acc = __builtin_amdgcn_wmma_f32_16x16x32_bf16(
            false, a, false, b, (short)0, acc, false, false);
    }

    // D layout: n = lane&15, m = reg + 8*(lane>>4)
    const int ncol = colBase + wcol + lm;
    const float bias = biascat[ncol];
#pragma unroll
    for (int r = 0; r < 8; r++) {
        int row = rowBase + wrow + r + 8 * lh;
        act[(size_t)row * COUT + ncol] = acc[r] + bias;
    }
}

// ---------------------------------------------------------------------------
// BN1/BNe batch stats over 96000 rows, channels 0..255 (two-stage, no atomics)
// ---------------------------------------------------------------------------
__global__ __launch_bounds__(256) void k_stats_ge_part(
    const float* __restrict__ act, float* __restrict__ psum, float* __restrict__ psq)
{
    int b = blockIdx.x;        // 250 blocks x 384 rows
    int c = threadIdx.x;       // channel
    float s = 0.0f, q = 0.0f;
    int r0 = b * 384;
    for (int r = r0; r < r0 + 384; r++) {
        float v = act[(size_t)r * COUT + c];
        s += v; q += v * v;
    }
    psum[b * 256 + c] = s;
    psq[b * 256 + c] = q;
}

__global__ __launch_bounds__(256) void k_stats_ge_final(
    const float* __restrict__ psum, const float* __restrict__ psq,
    const float* __restrict__ g1, const float* __restrict__ b1,
    const float* __restrict__ ge, const float* __restrict__ be,
    float* __restrict__ scale1, float* __restrict__ shift1)
{
    int c = threadIdx.x;
    float s = 0.0f, q = 0.0f;
    for (int i = 0; i < 250; i++) { s += psum[i * 256 + c]; q += psq[i * 256 + c]; }
    float mean = s / (float)NROWS;
    float var = q / (float)NROWS - mean * mean;
    float gamma = (c < 128) ? g1[c] : ge[c - 128];
    float beta  = (c < 128) ? b1[c] : be[c - 128];
    float sc = gamma * rsqrtf(var + BN_EPS);
    scale1[c] = sc;
    shift1[c] = beta - mean * sc;
}

// ---------------------------------------------------------------------------
// Per-atom moments of Pj / Pl  ->  S[n][{Su,Su2,Sv,Sv2}][c]
// ---------------------------------------------------------------------------
__global__ __launch_bounds__(128) void k_3body_rowstats(
    const float* __restrict__ act, float* __restrict__ S)
{
    int n = blockIdx.x, c = threadIdx.x;
    float su = 0, su2 = 0, sv = 0, sv2 = 0;
#pragma unroll
    for (int j = 0; j < MNBR; j++) {
        size_t r = (size_t)(n * MNBR + j) * COUT;
        float u = act[r + 256 + c];
        float v = act[r + 384 + c];
        su += u; su2 += u * u; sv += v; sv2 += v * v;
    }
    S[n * 512 + c]       = su;
    S[n * 512 + 128 + c] = su2;
    S[n * 512 + 256 + c] = sv;
    S[n * 512 + 384 + c] = sv2;
}

// Analytic BN3 stats over the never-materialized (N,M,M,128) tensor.
__global__ __launch_bounds__(256) void k_3body_bnstats(
    const float* __restrict__ S, const float* __restrict__ Pi,
    const float* __restrict__ g3, const float* __restrict__ b3g,
    float* __restrict__ scale3, float* __restrict__ shift3)
{
    __shared__ float rs[256], rq[256];
    int c = blockIdx.x;     // 128 channels
    int t = threadIdx.x;
    float sz = 0, sz2 = 0;
    for (int n = t; n < NATOM; n += 256) {
        float s   = Pi[n * 128 + c];   // includes b_3body
        float su  = S[n * 512 + c],       su2 = S[n * 512 + 128 + c];
        float sv  = S[n * 512 + 256 + c], sv2 = S[n * 512 + 384 + c];
        sz  += 144.0f * s + 12.0f * (su + sv);
        sz2 += 144.0f * s * s + 12.0f * (su2 + sv2) + 24.0f * s * (su + sv) + 2.0f * su * sv;
    }
    rs[t] = sz; rq[t] = sz2; __syncthreads();
    for (int o = 128; o > 0; o >>= 1) {
        if (t < o) { rs[t] += rs[t + o]; rq[t] += rq[t + o]; }
        __syncthreads();
    }
    if (t == 0) {
        const float cnt = (float)NATOM * 144.0f;
        float mean = rs[0] / cnt;
        float var = rq[0] / cnt - mean * mean;
        float sc = g3[c] * rsqrtf(var + BN_EPS);
        scale3[c] = sc;
        shift3[c] = b3g[c] - mean * sc;
    }
}

// ---------------------------------------------------------------------------
// two_body gate (sum over M) + edge output  nbr_out = nbr + sig(ef)*sp(ec)
// ---------------------------------------------------------------------------
__global__ __launch_bounds__(64) void k_twobody_edge(
    const float* __restrict__ act, const float* __restrict__ nbr,
    const float* __restrict__ scale1, const float* __restrict__ shift1,
    float* __restrict__ t2, float* __restrict__ out_nbr)
{
    int n = blockIdx.x, c = threadIdx.x;
    float s_gf = scale1[c],       h_gf = shift1[c];
    float s_gc = scale1[64 + c],  h_gc = shift1[64 + c];
    float s_ef = scale1[128 + c], h_ef = shift1[128 + c];
    float s_ec = scale1[192 + c], h_ec = shift1[192 + c];
    float acc = 0.0f;
#pragma unroll
    for (int j = 0; j < MNBR; j++) {
        size_t r = (size_t)(n * MNBR + j);
        const float* a = &act[r * COUT];
        float gf = a[c] * s_gf + h_gf;
        float gc = a[64 + c] * s_gc + h_gc;
        acc += sigmoid_f(gf) * softplus_f(gc);
        float ef = a[128 + c] * s_ef + h_ef;
        float ec = a[192 + c] * s_ec + h_ec;
        out_nbr[r * 64 + c] = nbr[r * 64 + c] + sigmoid_f(ef) * softplus_f(ec);
    }
    t2[n * 64 + c] = acc;
}

// ---------------------------------------------------------------------------
// three_body gate: 144 (j,l) pairs per (n,c) from LDS-staged scaled rows
// ---------------------------------------------------------------------------
__global__ __launch_bounds__(64) void k_threebody(
    const float* __restrict__ act, const float* __restrict__ Pi,
    const float* __restrict__ scale3, const float* __restrict__ shift3,
    float* __restrict__ t3)
{
    __shared__ float ujf[12 * 64], ujc[12 * 64], vlf[12 * 64], vlc[12 * 64];
    int n = blockIdx.x, c = threadIdx.x;
    float sf = scale3[c], sc2 = scale3[64 + c];
#pragma unroll
    for (int j = 0; j < MNBR; j++) {
        size_t r = (size_t)(n * MNBR + j) * COUT;
        ujf[j * 64 + c] = act[r + 256 + c] * sf;        // Pj filt
        ujc[j * 64 + c] = act[r + 320 + c] * sc2;       // Pj core
        vlf[j * 64 + c] = act[r + 384 + c] * sf;        // Pl filt
        vlc[j * 64 + c] = act[r + 448 + c] * sc2;       // Pl core
    }
    __syncthreads();
    float basef = Pi[n * 128 + c] * sf + shift3[c];
    float basec = Pi[n * 128 + 64 + c] * sc2 + shift3[64 + c];
    float acc = 0.0f;
    for (int j = 0; j < MNBR; j++) {
        float ff = basef + ujf[j * 64 + c];
        float cc = basec + ujc[j * 64 + c];
#pragma unroll
        for (int l = 0; l < MNBR; l++)
            acc += sigmoid_f(ff + vlf[l * 64 + c]) * softplus_f(cc + vlc[l * 64 + c]);
    }
    t3[n * 64 + c] = acc;
}

// ---------------------------------------------------------------------------
// BN2 stats over N per channel (64), then final atom output
// ---------------------------------------------------------------------------
__global__ __launch_bounds__(256) void k_bn2_stats(
    const float* __restrict__ t2, const float* __restrict__ t3,
    const float* __restrict__ g2, const float* __restrict__ b2,
    float* __restrict__ scale2, float* __restrict__ shift2)
{
    __shared__ float rs[256], rq[256];
    int t = threadIdx.x;
    int c = t & 63, s4 = t >> 6;
    float sm = 0, sq = 0;
    for (int n = s4; n < NATOM; n += 4) {
        float v = t2[n * 64 + c] + t3[n * 64 + c];
        sm += v; sq += v * v;
    }
    rs[t] = sm; rq[t] = sq; __syncthreads();
    if (t < 128) { rs[t] += rs[t + 128]; rq[t] += rq[t + 128]; } __syncthreads();
    if (t < 64) {
        rs[t] += rs[t + 64]; rq[t] += rq[t + 64];
        float mean = rs[t] / (float)NATOM;
        float var = rq[t] / (float)NATOM - mean * mean;
        float sc = g2[t] * rsqrtf(var + BN_EPS);
        scale2[t] = sc;
        shift2[t] = b2[t] - mean * sc;
    }
}

__global__ __launch_bounds__(256) void k_final(
    const float* __restrict__ atom_in, const float* __restrict__ t2,
    const float* __restrict__ t3, const float* __restrict__ scale2,
    const float* __restrict__ shift2, float* __restrict__ out_atom)
{
    int i = blockIdx.x * 256 + threadIdx.x;   // 512000
    int c = i & 63;
    float v = (t2[i] + t3[i]) * scale2[c] + shift2[c];
    out_atom[i] = softplus_f(atom_in[i] + v);
}

// ---------------------------------------------------------------------------
extern "C" void kernel_launch(void* const* d_in, const int* in_sizes, int n_in,
                              void* d_out, int out_size, void* d_ws, size_t ws_size,
                              hipStream_t stream)
{
    const float* atom_in = (const float*)d_in[0];
    const float* nbr     = (const float*)d_in[1];
    const int*   idx     = (const int*)d_in[2];
    const float* Wf      = (const float*)d_in[3];
    const float* bfull   = (const float*)d_in[4];
    const float* g1      = (const float*)d_in[5];
    const float* b1      = (const float*)d_in[6];
    const float* We      = (const float*)d_in[7];
    const float* bedge   = (const float*)d_in[8];
    const float* ge      = (const float*)d_in[9];
    const float* be      = (const float*)d_in[10];
    const float* W3      = (const float*)d_in[11];
    const float* b3      = (const float*)d_in[12];
    const float* g3      = (const float*)d_in[13];
    const float* b3b     = (const float*)d_in[14];
    const float* g2      = (const float*)d_in[15];
    const float* b2      = (const float*)d_in[16];
    (void)in_sizes; (void)n_in; (void)out_size; (void)ws_size;

    float* out_atom = (float*)d_out;             // (8000, 64)
    float* out_nbr  = out_atom + NATOM * AF;     // (8000, 12, 64)

    char* ws = (char*)d_ws;
    size_t off = 0;
    auto walloc = [&](size_t bytes) -> char* {
        char* p = ws + off;
        off = (off + bytes + 255) & ~(size_t)255;
        return p;
    };
    bf16_t* Wcat   = (bf16_t*)walloc((size_t)KTOT * COUT * sizeof(bf16_t));
    float* biascat = (float*)walloc(COUT * 4);
    float* act     = (float*)walloc((size_t)NROWS * COUT * 4);   // 196.6 MB
    float* Pi      = (float*)walloc((size_t)NATOM * 128 * 4);
    float* S       = (float*)walloc((size_t)NATOM * 512 * 4);
    float* t2      = (float*)walloc((size_t)NATOM * 64 * 4);
    float* t3      = (float*)walloc((size_t)NATOM * 64 * 4);
    float* psum    = (float*)walloc(250 * 256 * 4);
    float* psq     = (float*)walloc(250 * 256 * 4);
    float* scale1  = (float*)walloc(256 * 4);
    float* shift1  = (float*)walloc(256 * 4);
    float* scale3  = (float*)walloc(128 * 4);
    float* shift3  = (float*)walloc(128 * 4);
    float* scale2  = (float*)walloc(64 * 4);
    float* shift2  = (float*)walloc(64 * 4);

    hipLaunchKernelGGL(k_prep_weights, dim3(384), dim3(256), 0, stream,
                       Wf, We, W3, bfull, bedge, Wcat, biascat);
    hipLaunchKernelGGL(k_pi, dim3(NATOM), dim3(128), 0, stream, atom_in, W3, b3, Pi);
    hipLaunchKernelGGL(k_gemm, dim3(NROWS / 64, COUT / 32), dim3(256), 0, stream,
                       atom_in, nbr, idx, Wcat, biascat, act);
    hipLaunchKernelGGL(k_stats_ge_part, dim3(250), dim3(256), 0, stream, act, psum, psq);
    hipLaunchKernelGGL(k_stats_ge_final, dim3(1), dim3(256), 0, stream,
                       psum, psq, g1, b1, ge, be, scale1, shift1);
    hipLaunchKernelGGL(k_3body_rowstats, dim3(NATOM), dim3(128), 0, stream, act, S);
    hipLaunchKernelGGL(k_3body_bnstats, dim3(128), dim3(256), 0, stream,
                       S, Pi, g3, b3b, scale3, shift3);
    hipLaunchKernelGGL(k_twobody_edge, dim3(NATOM), dim3(64), 0, stream,
                       act, nbr, scale1, shift1, t2, out_nbr);
    hipLaunchKernelGGL(k_threebody, dim3(NATOM), dim3(64), 0, stream,
                       act, Pi, scale3, shift3, t3);
    hipLaunchKernelGGL(k_bn2_stats, dim3(1), dim3(256), 0, stream,
                       t2, t3, g2, b2, scale2, shift2);
    hipLaunchKernelGGL(k_final, dim3(NATOM * AF / 256), dim3(256), 0, stream,
                       atom_in, t2, t3, scale2, shift2, out_atom);
}
